// SelfAttention_52501680226519
// MI455X (gfx1250) — matrix operand order repaired and checked by reading, not compiled, not run on hardware
//
#include <hip/hip_runtime.h>

// ---------------------------------------------------------------------------
// CDNA5 (gfx1250) fused self-attention: bf16 WMMA everywhere, fp32 accumulate.
// - V produced pre-transposed ([B,H,D,S]): no LDS V-transpose, no barriers.
// - Softmax row-sums via WMMA against a ones matrix (matrix pipe).
// - Softmax row-max via DPP16 butterfly (VALU only).
// - K fragments software-pipelined one key-block ahead; 1/sqrt(d) folded into
//   the Q projection epilogue.
// ---------------------------------------------------------------------------

typedef unsigned short u16;
typedef __bf16 bf16;
typedef bf16  v16bf __attribute__((ext_vector_type(16)));
typedef float v8f   __attribute__((ext_vector_type(8)));

struct alignas(16) U4 { unsigned int x, y, z, w; };
struct alignas(16) F4 { float x, y, z, w; };

union FragU {
    U4    q[2];
    u16   u[16];
    v16bf v;
};

// fp32 -> bf16 with round-to-nearest-even
__device__ __forceinline__ u16 f2bf(float f) {
    unsigned u = __float_as_uint(f);
    u += 0x7FFFu + ((u >> 16) & 1u);
    return (u16)(u >> 16);
}

// DPP16 cross-lane move (within a 16-lane row), VALU-only
template <int CTRL>
__device__ __forceinline__ float dpp_xfer(float x) {
    return __int_as_float(__builtin_amdgcn_update_dpp(
        0, __float_as_int(x), CTRL, 0xF, 0xF, true));
}

// max-reduce across the 16 lanes of a half-wave: xor1, xor2, mirror8, mirror16
__device__ __forceinline__ float rowmax16(float x) {
    x = fmaxf(x, dpp_xfer<0xB1>(x));    // quad_perm(1,0,3,2)
    x = fmaxf(x, dpp_xfer<0x4E>(x));    // quad_perm(2,3,0,1)
    x = fmaxf(x, dpp_xfer<0x141>(x));   // row_half_mirror
    x = fmaxf(x, dpp_xfer<0x140>(x));   // row_mirror
    return x;
}

// B-fragment pattern: lane = column N, elements 0..7 = K[c0..c0+7], 8..15 = K[c0+16..c0+23], c0 = 8*(lane/16)
__device__ __forceinline__ v16bf load_frag16(const u16* p) {
    FragU f;
    f.q[0] = *(const U4*)(p);
    f.q[1] = *(const U4*)(p + 16);
    return f.v;
}

// A-fragment pattern per ISA: elements 0..7 = K[c0..c0+7], 8..15 = K[c0+16..c0+23]
// caller passes c0 = k0 + 8*(lane/16)
__device__ __forceinline__ v16bf load_afrag_bf(const u16* row, int c0) {
    FragU f;
    f.q[0] = *(const U4*)(row + c0);
    f.q[1] = *(const U4*)(row + c0 + 16);
    return f.v;
}

__device__ __forceinline__ v16bf load_afrag_f32(const float* row, int c0) {
    F4 a = *(const F4*)(row + c0);
    F4 b = *(const F4*)(row + c0 + 4);
    F4 c = *(const F4*)(row + c0 + 16);
    F4 d = *(const F4*)(row + c0 + 20);
    FragU f;
    f.u[0]  = f2bf(a.x); f.u[1]  = f2bf(a.y); f.u[2]  = f2bf(a.z); f.u[3]  = f2bf(a.w);
    f.u[4]  = f2bf(b.x); f.u[5]  = f2bf(b.y); f.u[6]  = f2bf(b.z); f.u[7]  = f2bf(b.w);
    f.u[8]  = f2bf(c.x); f.u[9]  = f2bf(c.y); f.u[10] = f2bf(c.z); f.u[11] = f2bf(c.w);
    f.u[12] = f2bf(d.x); f.u[13] = f2bf(d.y); f.u[14] = f2bf(d.z); f.u[15] = f2bf(d.w);
    return f.v;
}

__device__ __forceinline__ v8f wmma_bf16(v16bf a, v16bf b, v8f c) {
    return __builtin_amdgcn_wmma_f32_16x16x32_bf16(
        /*neg_a=*/false, a, /*neg_b=*/false, b,
        /*c_mod=*/(short)0, c, /*reuse_a=*/false, /*reuse_b=*/false);
}

// ---------------------------------------------------------------------------
// Stage 1: weight conversion fp32 -> bf16
// ---------------------------------------------------------------------------
__global__ __launch_bounds__(256) void cvt_f32_bf16(const float* __restrict__ src,
                                                    u16* __restrict__ dst, int n) {
    int i = blockIdx.x * 256 + threadIdx.x;
    if (i < n) dst[i] = f2bf(src[i]);
}

// ---------------------------------------------------------------------------
// Stage 2/4: GEMM  Y[M,N] = (A[M,K] * W[N,K]^T + bias[N]) * outScale
//   AF32:    A is fp32 (converted to bf16 in-register), else A is bf16
//   OUTMODE: 0 = fp32 row-major [M,N]
//            1 = bf16 split-head [B=4,H=16,S=2048,D=64]   (Q, K)
//            2 = bf16 transposed split-head [B,H,D=64,S=2048] (V)
// Block = 128 threads = 4 waves; wave tile = 16(M) x 64(N); WG tile = 64x64.
// ---------------------------------------------------------------------------
template <bool AF32, int OUTMODE>
__global__ __launch_bounds__(128) void gemm_wmma(const float* __restrict__ A32,
                                                 const u16* __restrict__ Abf,
                                                 const u16* __restrict__ W,
                                                 const float* __restrict__ bias,
                                                 float* __restrict__ outF,
                                                 u16* __restrict__ outH,
                                                 int M, int N, int Kdim,
                                                 float outScale) {
    const int tid  = threadIdx.x;
    const int wave = tid >> 5;
    const int lane = tid & 31;
    const int hl   = lane >> 4;   // which 16-lane half
    const int ln   = lane & 15;

    const int m0 = blockIdx.y * 64 + wave * 16;
    const int n0 = blockIdx.x * 64;

    v8f acc[4] = {};

    const size_t arow = (size_t)(m0 + ln) * Kdim;
    for (int k0 = 0; k0 < Kdim; k0 += 32) {
        v16bf af;
        if (AF32) af = load_afrag_f32(A32 + arow, k0 + 8 * hl);
        else      af = load_afrag_bf(Abf + arow, k0 + 8 * hl);
#pragma unroll
        for (int nt = 0; nt < 4; ++nt) {
            const int n = n0 + nt * 16 + ln;                 // lane's B column = W row
            v16bf bf_ = load_frag16(W + (size_t)n * Kdim + k0 + 8 * hl);
            acc[nt] = wmma_bf16(af, bf_, acc[nt]);
        }
    }

#pragma unroll
    for (int r = 0; r < 8; ++r) {
        const int m = m0 + r + 8 * hl;                       // C/D layout row
#pragma unroll
        for (int nt = 0; nt < 4; ++nt) {
            const int n = n0 + nt * 16 + ln;
            const float v = (acc[nt][r] + bias[n]) * outScale;
            if (OUTMODE == 1) {
                const int b = m >> 11, s = m & 2047;
                const int h = n >> 6,  d = n & 63;
                outH[(((size_t)b * 16 + h) * 2048 + s) * 64 + d] = f2bf(v);
            } else if (OUTMODE == 2) {
                const int b = m >> 11, s = m & 2047;
                const int h = n >> 6,  d = n & 63;
                outH[(((size_t)b * 16 + h) * 64 + d) * 2048 + s] = f2bf(v);
            } else {
                outF[(size_t)m * N + n] = v;
            }
        }
    }
}

// ---------------------------------------------------------------------------
// Stage 3: flash attention.
//   Q (pre-scaled by 1/sqrt(d)), K bf16 [B,H,S,D]; V bf16 [B,H,D,S]
//   -> Ctx bf16 [B,S,E].
// grid = (B*H = 64, S/64 = 32), block = 128 (4 independent waves, 16 queries
// each, no workgroup barriers). K fragments pipelined one key-block ahead.
// ---------------------------------------------------------------------------
__global__ __launch_bounds__(128) void attn_wmma(const u16* __restrict__ Q,
                                                 const u16* __restrict__ K,
                                                 const u16* __restrict__ Vt,
                                                 u16* __restrict__ Ctx) {
    const int S = 2048, D = 64;
    const int bh = blockIdx.x;
    const int b = bh >> 4, h = bh & 15;
    const int tid  = threadIdx.x;
    const int wave = tid >> 5;
    const int lane = tid & 31;
    const int hl   = lane >> 4;
    const int ln   = lane & 15;
    const int q0   = blockIdx.y * 64 + wave * 16;

    const u16* Qb = Q + (size_t)bh * S * D;
    const u16* Kb = K + (size_t)bh * S * D;

    __shared__ __align__(16) u16 Pl[4][16 * 32];    // per-wave P re-layout buffer

    // Q fragments (K-dim = d = 64 -> two 16x16x32 A fragments)
    const u16* qrow = Qb + (size_t)(q0 + ln) * D;
    const v16bf aq0 = load_afrag_bf(qrow, 0  + 8 * hl);
    const v16bf aq1 = load_afrag_bf(qrow, 32 + 8 * hl);

    // all-ones B fragment (bf16 1.0 = 0x3F80) for WMMA row-sum
    FragU of;
#pragma unroll
    for (int i = 0; i < 16; ++i) of.u[i] = 0x3F80u;
    const v16bf bones = of.v;

    // per-lane V row pointers: Vt row = d, keys contiguous (16B aligned chunks)
    const u16* vrow[4];
#pragma unroll
    for (int dt = 0; dt < 4; ++dt)
        vrow[dt] = Vt + ((size_t)bh * 64 + dt * 16 + ln) * (size_t)S + 8 * hl;

    // K fragment loader (B-fragment layout: lane = key column, d contiguous)
    const u16* krow_lo = Kb + (size_t)ln * D + 8 * hl;          // + kb*D
    const u16* krow_hi = Kb + (size_t)(16 + ln) * D + 8 * hl;   // + kb*D

    float mrun[8];
    v8f o[4] = {};
    v8f lacc = {};
#pragma unroll
    for (int r = 0; r < 8; ++r) mrun[r] = -__builtin_inff();

    // pipeline prologue: K fragments for block 0
    v16bf kcur[4];
    kcur[0] = load_frag16(krow_lo);
    kcur[1] = load_frag16(krow_lo + 32);
    kcur[2] = load_frag16(krow_hi);
    kcur[3] = load_frag16(krow_hi + 32);

    for (int kb = 0; kb < S; kb += 32) {
        // V fragments for this block (consumed after softmax: latency hidden)
        v16bf vcur[4];
#pragma unroll
        for (int dt = 0; dt < 4; ++dt)
            vcur[dt] = load_frag16(vrow[dt] + kb);

        // K fragments for the NEXT block (branchless wrap on final iteration)
        const int kbn = (kb + 32) & (S - 1);
        v16bf knxt[4];
        knxt[0] = load_frag16(krow_lo + (size_t)kbn * D);
        knxt[1] = load_frag16(krow_lo + (size_t)kbn * D + 32);
        knxt[2] = load_frag16(krow_hi + (size_t)kbn * D);
        knxt[3] = load_frag16(krow_hi + (size_t)kbn * D + 32);

        // prefetch two blocks ahead into caches
        if (kb + 64 < S) {
            __builtin_prefetch(Kb + (size_t)(kb + 64 + ln) * D, 0, 3);
            __builtin_prefetch(vrow[0] + kb + 64, 0, 3);
            __builtin_prefetch(vrow[2] + kb + 64, 0, 3);
        }

        // S = Q * K^T for 32 keys (Q pre-scaled by 1/sqrt(d))
        v8f s0 = {}, s1 = {};
        s0 = wmma_bf16(aq0, kcur[0], s0);
        s0 = wmma_bf16(aq1, kcur[1], s0);
        s1 = wmma_bf16(aq0, kcur[2], s1);
        s1 = wmma_bf16(aq1, kcur[3], s1);

        // online softmax; C/D layout: VGPR r -> row r+8*hl, cols across half-wave.
#pragma unroll
        for (int r = 0; r < 8; ++r) {
            const float mx = rowmax16(fmaxf(s0[r], s1[r]));
            const float nm = fmaxf(mrun[r], mx);
            const float a  = __expf(mrun[r] - nm);
            const float p0 = __expf(s0[r] - nm);
            const float p1 = __expf(s1[r] - nm);
            mrun[r] = nm;
            const int row = r + 8 * hl;
            Pl[wave][row * 32 + ln]      = f2bf(p0);
            Pl[wave][row * 32 + 16 + ln] = f2bf(p1);
            o[0][r] *= a; o[1][r] *= a; o[2][r] *= a; o[3][r] *= a;
            lacc[r] *= a;
        }

        // P (C-layout) -> A-fragment via per-wave LDS round trip (same-wave,
        // hardware DScnt ordering, no barrier needed)
        const u16* prow = &Pl[wave][ln * 32];
        const v16bf ap = load_afrag_bf(prow, 8 * hl);

        // row sums: lacc += P * ones (every column of result = rowsum)
        lacc = wmma_bf16(ap, bones, lacc);

        // O += P * V  (four 16-wide d tiles, V pre-transposed in global)
#pragma unroll
        for (int dt = 0; dt < 4; ++dt)
            o[dt] = wmma_bf16(ap, vcur[dt], o[dt]);

#pragma unroll
        for (int i = 0; i < 4; ++i) kcur[i] = knxt[i];
    }

    // normalize and store context in [B,S,E] (E index = h*64 + d)
#pragma unroll
    for (int r = 0; r < 8; ++r) {
        const float inv = 1.0f / lacc[r];
        const int m = q0 + r + 8 * hl;
        const size_t rowoff = ((size_t)b * 2048 + m) * 1024 + h * 64;
#pragma unroll
        for (int dt = 0; dt < 4; ++dt)
            Ctx[rowoff + dt * 16 + ln] = f2bf(o[dt][r] * inv);
    }
}

// ---------------------------------------------------------------------------
// Host-side launcher
// ---------------------------------------------------------------------------
extern "C" void kernel_launch(void* const* d_in, const int* in_sizes, int n_in,
                              void* d_out, int out_size, void* d_ws, size_t ws_size,
                              hipStream_t stream) {
    (void)in_sizes; (void)n_in; (void)out_size; (void)ws_size;

    const float* query = (const float*)d_in[0];
    const float* key   = (const float*)d_in[1];
    const float* value = (const float*)d_in[2];
    const float* Wq    = (const float*)d_in[3];
    const float* bq    = (const float*)d_in[4];
    const float* Wk    = (const float*)d_in[5];
    const float* bk    = (const float*)d_in[6];
    const float* Wv    = (const float*)d_in[7];
    const float* bv    = (const float*)d_in[8];
    const float* Wo    = (const float*)d_in[9];
    const float* bo    = (const float*)d_in[10];

    const int M = 4 * 2048;        // B*S
    const int N = 1024;            // E
    const int Kd = 1024;           // E
    const int WN = 1024 * 1024;    // weight elements

    // workspace layout (bf16 buffers), ~72 MB total
    char* ws = (char*)d_ws;
    u16* WqB = (u16*)(ws);
    u16* WkB = (u16*)(ws + 2u * 1024 * 1024);
    u16* WvB = (u16*)(ws + 4u * 1024 * 1024);
    u16* WoB = (u16*)(ws + 6u * 1024 * 1024);
    u16* Qb  = (u16*)(ws + 8u * 1024 * 1024);
    u16* Kb  = (u16*)(ws + 24u * 1024 * 1024);
    u16* Vb  = (u16*)(ws + 40u * 1024 * 1024);   // [B,H,D,S] transposed
    u16* Ctx = (u16*)(ws + 56u * 1024 * 1024);

    // 1) weights -> bf16
    const int cb = (WN + 255) / 256;
    cvt_f32_bf16<<<cb, 256, 0, stream>>>(Wq, WqB, WN);
    cvt_f32_bf16<<<cb, 256, 0, stream>>>(Wk, WkB, WN);
    cvt_f32_bf16<<<cb, 256, 0, stream>>>(Wv, WvB, WN);
    cvt_f32_bf16<<<cb, 256, 0, stream>>>(Wo, WoB, WN);

    // 2) QKV projections (Q pre-scaled by 1/sqrt(64) = 0.125)
    dim3 gg(N / 64, M / 64);
    gemm_wmma<true, 1><<<gg, 128, 0, stream>>>(query, nullptr, WqB, bq, nullptr, Qb, M, N, Kd, 0.125f);
    gemm_wmma<true, 1><<<gg, 128, 0, stream>>>(key,   nullptr, WkB, bk, nullptr, Kb, M, N, Kd, 1.0f);
    gemm_wmma<true, 2><<<gg, 128, 0, stream>>>(value, nullptr, WvB, bv, nullptr, Vb, M, N, Kd, 1.0f);

    // 3) flash attention -> Ctx bf16 [B,S,E]
    attn_wmma<<<dim3(64, 32), 128, 0, stream>>>(Qb, Kb, Vb, Ctx);

    // 4) output projection -> fp32 d_out
    gemm_wmma<false, 0><<<gg, 128, 0, stream>>>(nullptr, Ctx, WoB, bo,
                                                (float*)d_out, nullptr, M, N, Kd, 1.0f);
}
